// Barrier_Net_39599598469458
// MI455X (gfx1250) — matrix operand order repaired and checked by reading, compiled, not run on hardware
//
#include <hip/hip_runtime.h>
#include <math.h>
#include <stdint.h>

typedef __attribute__((ext_vector_type(16))) _Float16 v16h;
typedef __attribute__((ext_vector_type(8)))  float    v8f;

#define BN       100000
#define NNB      16
#define NOB      8
#define SDF      4
#define HID      64
#define PHIO     16
#define D_OBS    85     // 1 + (NN+1)*SD + 2*NO
#define NB_OFF   5      // 1 + SD
#define OB_OFF   69     // 1 + (NN+1)*SD
#define HP       72     // padded LDS row (64 + 8)
#define XS_N     (16 * D_OBS)  // staged dwords per wave (1360)

__device__ __forceinline__ int imin(int a, int b) { return a < b ? a : b; }

__device__ __forceinline__ v8f zero8() {
  v8f z;
#pragma unroll
  for (int i = 0; i < 8; ++i) z[i] = 0.0f;
  return z;
}

__device__ __forceinline__ v16h zero16() {
  v16h z;
#pragma unroll
  for (int i = 0; i < 16; ++i) z[i] = (_Float16)0.0f;
  return z;
}

__device__ __forceinline__ v8f wmma16(v16h a, v16h b, v8f c) {
  // D = A(16x32 f16) * B(32x16 f16) + C(16x16 f32)
  return __builtin_amdgcn_wmma_f32_16x16x32_f16(false, a, false, b, (short)0, c,
                                                false, false);
}

// Same-wave LDS ordering fence: stop compiler reordering around wave-shared
// staging buffers (hardware keeps same-wave DS ops in order).
__device__ __forceinline__ void wave_sync() {
  __builtin_amdgcn_fence(__ATOMIC_RELEASE, "wavefront");
  __builtin_amdgcn_wave_barrier();
  __builtin_amdgcn_fence(__ATOMIC_ACQUIRE, "wavefront");
}

// Build a WMMA B operand (32x16, f16) from row-major f32 W[Krows x Ncols] (ld),
// taking K rows [k0,k0+32) and N cols [n0,n0+16), zero padded.
// B operand in this code's contiguous k order (same as A): element i of lane L holds (K = k0 + (L&16) + i,
// N = n0 + (L&15)).
__device__ __forceinline__ v16h make_btile(const float* __restrict__ W, int ld,
                                           int Krows, int Ncols, int k0, int n0,
                                           int lane) {
  v16h b;
  const int n  = n0 + (lane & 15);
  const int kb = k0 + ((lane & 16) ? 16 : 0);
#pragma unroll
  for (int i = 0; i < 16; ++i) {
    const int k = kb + i;
    float v = (k < Krows && n < Ncols) ? W[k * ld + n] : 0.0f;
    b[i] = (_Float16)v;
  }
  return b;
}

// Load a WMMA A operand (16x32, f16) from an LDS row-major [16][HP] buffer.
// A operand in this code's contiguous k order (same as B): lane L, element i -> (M = L&15,
// K = c32 + 2*kh + i), kh = (L&16) ? 8 : 0.
__device__ __forceinline__ v16h load_atile_lds(const _Float16* row, int c32, int kh) {
  v16h a;
#pragma unroll
  for (int i = 0; i < 16; ++i) {
    const int k = c32 + 2 * kh + i;
    a[i] = row[k];
  }
  return a;
}

__global__ __launch_bounds__(256) void barrier_net_wmma(
    const float* __restrict__ x,
    const float* __restrict__ phi_w1, const float* __restrict__ phi_b1,
    const float* __restrict__ phi_w2, const float* __restrict__ phi_b2,
    const float* __restrict__ obs_w1, const float* __restrict__ obs_b1,
    const float* __restrict__ obs_w2, const float* __restrict__ obs_b2,
    const float* __restrict__ rho_w1, const float* __restrict__ rho_b1,
    const float* __restrict__ rho_w2, const float* __restrict__ rho_b2,
    const float* __restrict__ psi_w1, const float* __restrict__ psi_b1,
    const float* __restrict__ psi_w2, const float* __restrict__ psi_b2,
    float* __restrict__ out) {
  __shared__ float    xs[8][XS_N];       // per-wave async-staged input rows
  __shared__ _Float16 hstage[8][16][HP]; // per-wave hidden staging (16 x 64 f16)
  __shared__ float    estage[8][16][2];  // per-wave 2-wide staging

  const int  lane = threadIdx.x & 31;
  const int  wv   = threadIdx.x >> 5;
  const int  m    = lane & 15;
  const bool hiH  = (lane & 16) != 0;
  const int  kh   = hiH ? 8 : 0;
  const int  s0   = (blockIdx.x * (blockDim.x >> 5) + wv) * 16;  // 16 samples/wave
  const int  Bm1  = BN - 1;

  _Float16 (*hrow)[HP] = hstage[wv];
  const float* xsw = &xs[wv][0];

  // ---- async DMA: stage this wave's 16 input rows (16*85 f32) into LDS ----
  {
    const uint32_t ldsbase = (uint32_t)(uintptr_t)&xs[wv][0];  // LDS byte addr
    const size_t   gbase   = (size_t)s0 * D_OBS;
    const size_t   gmax    = (size_t)BN * D_OBS - 1;
    for (int i = lane; i < XS_N; i += 32) {
      size_t gi = gbase + (size_t)i;
      if (gi > gmax) gi = gmax;                  // clamp (tail), stores masked later
      const uint32_t voff  = (uint32_t)(gi * 4u);  // byte offset from x
      const uint32_t laddr = ldsbase + (uint32_t)(i * 4);
      asm volatile("global_load_async_to_lds_b32 %0, %1, %2"
                   :
                   : "v"(laddr), "v"(voff), "s"(x)
                   : "memory");
    }
  }

  // ---------------- weight operand tiles (B-layout, f16) ----------------
  // (global loads overlap the async DMA above)
  v16h pW1[4], pW2[2], oW1[4], oW2[2];
#pragma unroll
  for (int j = 0; j < 4; ++j) pW1[j] = make_btile(phi_w1, HID, SDF, HID, 0, 16 * j, lane);
#pragma unroll
  for (int c = 0; c < 2; ++c) pW2[c] = make_btile(phi_w2, PHIO, HID, PHIO, 32 * c, 0, lane);
#pragma unroll
  for (int j = 0; j < 4; ++j) oW1[j] = make_btile(obs_w1, HID, 2, HID, 0, 16 * j, lane);
#pragma unroll
  for (int c = 0; c < 2; ++c) oW2[c] = make_btile(obs_w2, PHIO, HID, PHIO, 32 * c, 0, lane);

  float pb1[4], ob1[4];
#pragma unroll
  for (int j = 0; j < 4; ++j) { pb1[j] = phi_b1[16 * j + m]; ob1[j] = obs_b1[16 * j + m]; }
  const float pb2 = phi_b2[m];
  const float ob2 = obs_b2[m];

  // PS tile accumulator: C-layout (16 samples x 16 features), f32.
  float psA[8];
#pragma unroll
  for (int r = 0; r < 8; ++r) psA[r] = 0.0f;

  // wait for staged inputs
  asm volatile("s_wait_asynccnt 0" ::: "memory");
  wave_sync();

  // ---------------- phi: 16 tiles, one sample each (rows = neighbors) ----
#pragma unroll
  for (int t = 0; t < 16; ++t) {
    v16h a = zero16();
    if (!hiH) {  // low lanes carry K=0..3 (features); high lanes are zero pad
      const float* nbp = xsw + t * D_OBS + NB_OFF + SDF * m;  // neighbor m
      a[0] = (_Float16)nbp[0]; a[1] = (_Float16)nbp[1];
      a[2] = (_Float16)nbp[2]; a[3] = (_Float16)nbp[3];
    }

#pragma unroll
    for (int j = 0; j < 4; ++j) {
      v8f hv = wmma16(a, pW1[j], zero8());
#pragma unroll
      for (int r = 0; r < 8; ++r) {
        float v = hv[r] + pb1[j];
        v = v > 0.0f ? v : 0.0f;  // relu
        hrow[r + kh][16 * j + m] = (_Float16)v;
      }
    }
    wave_sync();
    v8f p = zero8();
#pragma unroll
    for (int c = 0; c < 2; ++c) {
      v16h a2 = load_atile_lds(&hrow[m][0], 32 * c, kh);
      p = wmma16(a2, pW2[c], p);
    }
    wave_sync();

    float rs = 0.0f;
#pragma unroll
    for (int r = 0; r < 8; ++r) rs += p[r];
    rs += __shfl_xor(rs, 16, 32);       // sum over all 16 neighbor rows
    const float ps = rs + 16.0f * pb2;  // + NN * phi_b2
    psA[t & 7] += ((lane < 16) == (t < 8)) ? ps : 0.0f;
  }

  // ---------------- obs: 8 tiles, two samples each (rows = obstacles) ----
#pragma unroll
  for (int t = 0; t < 8; ++t) {
    v16h a = zero16();
    if (!hiH) {  // rows 0-7: obstacles of sample 2t; rows 8-15: sample 2t+1
      const int    sl  = 2 * t + ((m < 8) ? 0 : 1);  // local sample index
      const float* obp = xsw + sl * D_OBS + OB_OFF + 2 * (m & 7);
      a[0] = (_Float16)obp[0]; a[1] = (_Float16)obp[1];
    }

#pragma unroll
    for (int j = 0; j < 4; ++j) {
      v8f hv = wmma16(a, oW1[j], zero8());
#pragma unroll
      for (int r = 0; r < 8; ++r) {
        float v = hv[r] + ob1[j];
        v = v > 0.0f ? v : 0.0f;
        hrow[r + kh][16 * j + m] = (_Float16)v;
      }
    }
    wave_sync();
    v8f p = zero8();
#pragma unroll
    for (int c = 0; c < 2; ++c) {
      v16h a2 = load_atile_lds(&hrow[m][0], 32 * c, kh);
      p = wmma16(a2, oW2[c], p);
    }
    wave_sync();

    float u = 0.0f;
#pragma unroll
    for (int r = 0; r < 8; ++r) u += p[r];
    const float v   = __shfl_xor(u, 16, 32);
    const float sLo = hiH ? v : u;  // obs sum of sample 2t   (col = m)
    const float sHi = hiH ? u : v;  // obs sum of sample 2t+1
    const float b2t = 8.0f * ob2;   // + NO * obs_b2
    const int r0 = 2 * t, r1 = 2 * t + 1;
    psA[r0 & 7] += ((lane < 16) == (r0 < 8)) ? (sLo + b2t) : 0.0f;
    psA[r1 & 7] += ((lane < 16) == (r1 < 8)) ? (sHi + b2t) : 0.0f;
  }

  // ---------------- rho / psi weight tiles ----------------
  v16h rW1[4], rW2[2], sW1[4], sW2[2];
#pragma unroll
  for (int j = 0; j < 4; ++j) rW1[j] = make_btile(rho_w1, HID, PHIO, HID, 0, 16 * j, lane);
#pragma unroll
  for (int c = 0; c < 2; ++c) rW2[c] = make_btile(rho_w2, 2, HID, 2, 32 * c, 0, lane);
#pragma unroll
  for (int j = 0; j < 4; ++j) sW1[j] = make_btile(psi_w1, HID, 4, HID, 0, 16 * j, lane);
#pragma unroll
  for (int c = 0; c < 2; ++c) sW2[c] = make_btile(psi_w2, 2, HID, 2, 32 * c, 0, lane);
  float rb1[4], sb1[4];
#pragma unroll
  for (int j = 0; j < 4; ++j) { rb1[j] = rho_b1[16 * j + m]; sb1[j] = psi_b1[16 * j + m]; }

  // ---------------- rho: 16 samples x 16 features -> 64 -> 2 -------------
  wave_sync();
#pragma unroll
  for (int r = 0; r < 8; ++r) hrow[r + kh][m] = (_Float16)psA[r];  // stage PS (16x16)
  wave_sync();
  v16h aR = zero16();
#pragma unroll
  for (int i = 0; i < 16; ++i) aR[i] = hiH ? (_Float16)0.0f : hrow[m][i];  // K=0..15, pad K>=16
  wave_sync();

#pragma unroll
  for (int j = 0; j < 4; ++j) {
    v8f hv = wmma16(aR, rW1[j], zero8());
#pragma unroll
    for (int r = 0; r < 8; ++r) {
      float v = hv[r] + rb1[j];
      v = v > 0.0f ? v : 0.0f;
      hrow[r + kh][16 * j + m] = (_Float16)v;
    }
  }
  wave_sync();
  v8f rAcc = zero8();
#pragma unroll
  for (int c = 0; c < 2; ++c) {
    v16h a2 = load_atile_lds(&hrow[m][0], 32 * c, kh);
    rAcc = wmma16(a2, rW2[c], rAcc);
  }
  wave_sync();
  if (m < 2) {  // stage rho outputs (cols 0,1) + bias
#pragma unroll
    for (int r = 0; r < 8; ++r) estage[wv][r + kh][m] = rAcc[r] + rho_b2[m];
  }
  wave_sync();

  // ---------------- psi: [rho0, rho1, g0, g1] -> 64 -> 2 -----------------
  const float r0v = estage[wv][m][0];
  const float r1v = estage[wv][m][1];
  const float g0  = xsw[m * D_OBS + 0];
  const float g1  = xsw[m * D_OBS + 1];

  v16h aP = zero16();
  if (!hiH) {
    aP[0] = (_Float16)r0v; aP[1] = (_Float16)r1v;
    aP[2] = (_Float16)g0;  aP[3] = (_Float16)g1;
  }

#pragma unroll
  for (int j = 0; j < 4; ++j) {
    v8f hv = wmma16(aP, sW1[j], zero8());
#pragma unroll
    for (int r = 0; r < 8; ++r) {
      float v = hv[r] + sb1[j];
      v = v > 0.0f ? v : 0.0f;
      hrow[r + kh][16 * j + m] = (_Float16)v;
    }
  }
  wave_sync();
  v8f eAcc = zero8();
#pragma unroll
  for (int c = 0; c < 2; ++c) {
    v16h a2 = load_atile_lds(&hrow[m][0], 32 * c, kh);
    eAcc = wmma16(a2, sW2[c], eAcc);
  }
  wave_sync();
  if (m < 2) {  // empty = tanh(e + psi_b2)
#pragma unroll
    for (int r = 0; r < 8; ++r) estage[wv][r + kh][m] = tanhf(eAcc[r] + psi_b2[m]);
  }
  wave_sync();

  // ---------------- barrier term + output (lanes 0..15, one sample each) --
  if (!hiH) {
    const int s = s0 + m;
    const float* xr = xsw + m * D_OBS;
    float bx = 0.0f, by = 0.0f;
#pragma unroll
    for (int j = 0; j < NNB; ++j) {
      const float p0 = -xr[NB_OFF + SDF * j + 0];
      const float p1 = -xr[NB_OFF + SDF * j + 1];
      const float h  = sqrtf(p0 * p0 + p1 * p1) - 0.2f;  // DS
      const float w  = 0.01f / h;                        // B_GAMMA / h^1
      bx += w * p0;
      by += w * p1;
    }
    const float e0 = estage[wv][m][0];
    const float e1 = estage[wv][m][1];
    if (s < BN) {
      out[2 * (size_t)s + 0] = 2.0f * tanhf(e0 + bx);
      out[2 * (size_t)s + 1] = 2.0f * tanhf(e1 + by);
    }
  }
  (void)Bm1;
}

extern "C" void kernel_launch(void* const* d_in, const int* in_sizes, int n_in,
                              void* d_out, int out_size, void* d_ws, size_t ws_size,
                              hipStream_t stream) {
  (void)in_sizes; (void)n_in; (void)out_size; (void)d_ws; (void)ws_size;
  const float* x      = (const float*)d_in[0];
  const float* phi_w1 = (const float*)d_in[1];
  const float* phi_b1 = (const float*)d_in[2];
  const float* phi_w2 = (const float*)d_in[3];
  const float* phi_b2 = (const float*)d_in[4];
  const float* obs_w1 = (const float*)d_in[5];
  const float* obs_b1 = (const float*)d_in[6];
  const float* obs_w2 = (const float*)d_in[7];
  const float* obs_b2 = (const float*)d_in[8];
  const float* rho_w1 = (const float*)d_in[9];
  const float* rho_b1 = (const float*)d_in[10];
  const float* rho_w2 = (const float*)d_in[11];
  const float* rho_b2 = (const float*)d_in[12];
  const float* psi_w1 = (const float*)d_in[13];
  const float* psi_b1 = (const float*)d_in[14];
  const float* psi_w2 = (const float*)d_in[15];
  const float* psi_b2 = (const float*)d_in[16];
  float* out = (float*)d_out;

  const int samplesPerBlock = 128;  // 8 waves * 16 samples
  const int grid = (BN + samplesPerBlock - 1) / samplesPerBlock;
  barrier_net_wmma<<<grid, 256, 0, stream>>>(
      x, phi_w1, phi_b1, phi_w2, phi_b2, obs_w1, obs_b1, obs_w2, obs_b2,
      rho_w1, rho_b1, rho_w2, rho_b2, psi_w1, psi_b1, psi_w2, psi_b2, out);
}
